// CKTGNN_17867063951410
// MI455X (gfx1250) — compile-verified, matrix operand order checked by reading, not run on hardware
//
#include <hip/hip_runtime.h>

// ---------------- CDNA5 WMMA types ----------------
typedef __attribute__((ext_vector_type(16))) __bf16 v16bf;
typedef __attribute__((ext_vector_type(8)))  float  v8f;

union ABu { v16bf v; uint4 u[2]; };

// ---------------- sizes ----------------
#define MAXN 12
#define PP   9
#define NT_  10
#define HID_ 301
#define BATCH 4096
#define G3   903          // 3*HID
#define NPAD_G 912        // 57 * 16
#define NT_G   57
#define KSTEPS 10         // K padded to 320 = 10*32
#define KPAD   320
#define NT_GM  19         // 301 -> 304
#define GMST   304
#define NT_FC  7          // 112 -> 112
#define FRAG   512        // bf16 elems per 32x16 B tile

// fragment regions in d_ws (bf16 units)
#define WHH_OFF 0
#define WHH_SZ  (NT_G * KSTEPS * FRAG)      // 291840
#define WG_OFF  (WHH_OFF + WHH_SZ)
#define WG_SZ   (NT_GM * KSTEPS * FRAG)     // 97280
#define WM_OFF  (WG_OFF + WG_SZ)
#define WFC_OFF (WM_OFF + WG_SZ)
#define WFC_SZ  (NT_FC * KSTEPS * FRAG)     // 35840
#define FRAG_TOTAL (WFC_OFF + WFC_SZ)       // 522240 bf16 ~ 1.02 MB

__device__ __forceinline__ unsigned short f2bf(float f) {
    unsigned u = __builtin_bit_cast(unsigned, f);
    unsigned r = (u + 0x7fffu + ((u >> 16) & 1u)) >> 16;
    return (unsigned short)r;
}
__device__ __forceinline__ float bf2f(unsigned short s) {
    return __builtin_bit_cast(float, ((unsigned)s) << 16);
}
__device__ __forceinline__ float sigm(float x) { return 1.f / (1.f + __expf(-x)); }
__device__ __forceinline__ float tanh_f(float x) {
    float e2 = __expf(-2.f * x);
    return (1.f - e2) / (1.f + e2);
}

// ---------------------------------------------------------------------------
// Prep: swizzle all weight matrices into bf16 B-fragments (32x16 K-major
// tiles, per-lane 16 contiguous bf16 = one 32B load in the GEMM hot loop).
// B[k][n] = W[n][k] (B = W^T), zero-padded so A padding never matters.
// ---------------------------------------------------------------------------
__global__ void prep_frags(const float* __restrict__ Whh,
                           const float* __restrict__ Wg,
                           const float* __restrict__ Wm,
                           const float* __restrict__ Wfc1,
                           const float* __restrict__ Wfc2,
                           unsigned short* __restrict__ ws) {
    int idx = blockIdx.x * blockDim.x + threadIdx.x;
    if (idx >= FRAG_TOTAL) return;
    int local = idx;
    const float* W = nullptr;
    int Nr = 0, Kc = 0, fc = 0;
    if (local < WHH_SZ)                       { W = Whh; Nr = G3;   Kc = HID_; }
    else if ((local -= WHH_SZ) < WG_SZ)       { W = Wg;  Nr = HID_; Kc = 310;  }
    else if ((local -= WG_SZ) < WG_SZ)        { W = Wm;  Nr = HID_; Kc = 310;  }
    else                                      { local -= WG_SZ; fc = 1; Kc = 309; }
    int slot  = local & 15;
    int lane  = (local >> 4) & 31;
    int tk    = local >> 9;            // ntile*KSTEPS + kstep (all regions KSTEPS=10)
    int kstep = tk % KSTEPS;
    int ntile = tk / KSTEPS;
    // CDNA5 16-bit B 32x16 layout (mirror of A layout, M<->N):
    int k = kstep * 32 + (slot & 7) + ((slot >= 8) ? 16 : 0) + ((lane >= 16) ? 8 : 0);
    int n = ntile * 16 + (lane & 15);
    float val = 0.f;
    if (!fc) {
        if (n < Nr && k < Kc) val = W[(size_t)n * Kc + k];
    } else {
        if (n < 112 && k < 309)
            val = (n < 56) ? Wfc1[(size_t)n * 309 + k] : Wfc2[(size_t)(n - 56) * 309 + k];
    }
    ws[idx] = f2bf(val);
}

// ---------------------------------------------------------------------------
// Main: one workgroup (8 wave32) owns 16 batch rows and runs the entire
// 12-step DAG-GRU scan in LDS, using v_wmma_f32_16x16x32_bf16 for every GEMM.
// ---------------------------------------------------------------------------
__global__ __launch_bounds__(256, 1)
void cktgnn_kernel(const int* __restrict__ types, const int* __restrict__ paths,
                   const int* __restrict__ adj_raw, const float* __restrict__ feats,
                   const float* __restrict__ W_ih, const float* __restrict__ b_ih,
                   const float* __restrict__ b_hh, const float* __restrict__ bg,
                   const float* __restrict__ W_df1, const float* __restrict__ b_df1,
                   const float* __restrict__ W_df2, const float* __restrict__ b_df2,
                   const float* __restrict__ b_fc1, const float* __restrict__ b_fc2,
                   const unsigned short* __restrict__ frags,
                   float* __restrict__ out) {
    __shared__ __attribute__((aligned(16))) float          g_s[16 * NPAD_G];      // gate / gated / fc GEMM output
    __shared__ __attribute__((aligned(16))) unsigned short psA[16 * KPAD];        // A-staging (bf16)
    __shared__ __attribute__((aligned(16))) float          h_s[16 * HID_];        // current hidden
    __shared__ __attribute__((aligned(16))) float          hin_s[16 * HID_];      // aggregated message
    __shared__ __attribute__((aligned(16))) unsigned short gated_s[MAXN * 16 * GMST]; // cached gated msgs (bf16)
    __shared__ float adj_s[16 * MAXN * MAXN];
    __shared__ int   types_s[16 * MAXN];
    __shared__ int   paths_s[16 * MAXN];
    __shared__ float feats_s[16 * MAXN * 3];

    const int tid  = threadIdx.x;
    const int lane = tid & 31;
    const int wave = tid >> 5;
    const int r0   = blockIdx.x * 16;

    // Warm L2 with the weight fragments (gfx1250 global_prefetch_b8).
    __builtin_prefetch(frags + (size_t)tid * (FRAG_TOTAL / 256), 0, 1);

    // ---- stage per-tile inputs ----
    for (int i = tid; i < 16 * MAXN; i += 256) {
        int row = i / MAXN, v = i % MAXN;
        types_s[i] = types[(size_t)(r0 + row) * MAXN + v];
        paths_s[i] = paths[(size_t)(r0 + row) * MAXN + v];
    }
    for (int i = tid; i < 16 * MAXN * MAXN; i += 256) {
        int row = i / (MAXN * MAXN), rem = i % (MAXN * MAXN);
        int v = rem / MAXN, u = rem % MAXN;
        adj_s[i] = (u < v) ? (float)adj_raw[(size_t)(r0 + row) * MAXN * MAXN + v * MAXN + u] : 0.f;
    }
    for (int i = tid; i < 16 * MAXN * 3; i += 256)
        feats_s[i] = feats[(size_t)(r0 + i / 36) * 36 + (i % 36)];
    for (int i = tid; i < 16 * NPAD_G; i += 256) g_s[i] = 0.f;     // gh == 0 at step 0
    for (int i = tid; i < 16 * HID_;   i += 256) hin_s[i] = 0.f;   // h_in == 0 at step 0
    for (int i = tid; i < 16 * KPAD;   i += 256) psA[i] = 0;
    __syncthreads();

    for (int v = 0; v < MAXN; ++v) {
        // ---- gh = h_in @ W_hh^T  (M=16, K=320, N=912) ----
        if (v > 0) {
            for (int t = wave; t < NT_G; t += 8) {
                v8f acc = {};
                const unsigned short* bp = frags + WHH_OFF + (size_t)t * KSTEPS * FRAG;
                #pragma unroll
                for (int ks = 0; ks < KSTEPS; ++ks) {
                    ABu a, b;
                    int abase = (lane & 15) * KPAD + ks * 32 + ((lane >= 16) ? 8 : 0);
                    a.u[0] = *(const uint4*)&psA[abase];
                    a.u[1] = *(const uint4*)&psA[abase + 16];
                    const unsigned short* bl = bp + ks * FRAG + lane * 16;
                    b.u[0] = *(const uint4*)&bl[0];
                    b.u[1] = *(const uint4*)&bl[8];
                    acc = __builtin_amdgcn_wmma_f32_16x16x32_bf16(
                        false, a.v, false, b.v, (short)0, acc, false, false);
                }
                int orow = (lane >= 16) ? 8 : 0, ocol = t * 16 + (lane & 15);
                #pragma unroll
                for (int r = 0; r < 8; ++r) g_s[(orow + r) * NPAD_G + ocol] = acc[r];
            }
        }
        __syncthreads();

        // ---- GRU cell (gi from 2-hot X: two scalar weight loads per elem) ----
        for (int i = tid; i < 16 * HID_; i += 256) {
            int row = i / HID_, j = i % HID_;
            int tpe = types_s[row * MAXN + v], pth = paths_s[row * MAXN + v];
            const float* wr = W_ih + (size_t)j * 19;
            const float* wz = W_ih + (size_t)(j + HID_) * 19;
            const float* wn = W_ih + (size_t)(j + 2 * HID_) * 19;
            float ir  = wr[tpe] + wr[NT_ + pth] + b_ih[j];
            float iz  = wz[tpe] + wz[NT_ + pth] + b_ih[j + HID_];
            float inn = wn[tpe] + wn[NT_ + pth] + b_ih[j + 2 * HID_];
            float hr = g_s[row * NPAD_G + j]            + b_hh[j];
            float hz = g_s[row * NPAD_G + HID_ + j]     + b_hh[j + HID_];
            float hn = g_s[row * NPAD_G + 2 * HID_ + j] + b_hh[j + 2 * HID_];
            float r = sigm(ir + hr);
            float z = sigm(iz + hz);
            float nn = tanh_f(inn + r * hn);
            float h = (1.f - z) * nn + z * hin_s[i];
            h_s[i] = h;
            psA[row * KPAD + j] = f2bf(h);
        }
        for (int i = tid; i < 16 * PP; i += 256) {   // X_pos one-hot -> ps[301..309]
            int row = i / PP, p = i % PP;
            psA[row * KPAD + HID_ + p] = (paths_s[row * MAXN + v] == p) ? f2bf(1.f) : 0;
        }
        __syncthreads();

        if (v < MAXN - 1) {
            // ---- gated_v = sigmoid(ps@Wg^T+bg) * (ps@Wm^T), computed ONCE ----
            for (int t = wave; t < 2 * NT_GM; t += 8) {
                int m = (t < NT_GM) ? 0 : 1;
                int tt = m ? t - NT_GM : t;
                const unsigned short* bp =
                    frags + (m ? WM_OFF : WG_OFF) + (size_t)tt * KSTEPS * FRAG;
                v8f acc = {};
                #pragma unroll
                for (int ks = 0; ks < KSTEPS; ++ks) {
                    ABu a, b;
                    int abase = (lane & 15) * KPAD + ks * 32 + ((lane >= 16) ? 8 : 0);
                    a.u[0] = *(const uint4*)&psA[abase];
                    a.u[1] = *(const uint4*)&psA[abase + 16];
                    const unsigned short* bl = bp + ks * FRAG + lane * 16;
                    b.u[0] = *(const uint4*)&bl[0];
                    b.u[1] = *(const uint4*)&bl[8];
                    acc = __builtin_amdgcn_wmma_f32_16x16x32_bf16(
                        false, a.v, false, b.v, (short)0, acc, false, false);
                }
                int orow = (lane >= 16) ? 8 : 0;
                int ocol = m * GMST + tt * 16 + (lane & 15);
                #pragma unroll
                for (int r = 0; r < 8; ++r) g_s[(orow + r) * NPAD_G + ocol] = acc[r];
            }
            __syncthreads();
            for (int i = tid; i < 16 * HID_; i += 256) {
                int row = i / HID_, j = i % HID_;
                float a = g_s[row * NPAD_G + j] + bg[j];
                float mm = g_s[row * NPAD_G + GMST + j];
                gated_s[(v * 16 + row) * GMST + j] = f2bf(sigm(a) * mm);
            }
            __syncthreads();
            // ---- h_in for step v+1: adjacency-weighted sum of cached gated ----
            for (int i = tid; i < 16 * HID_; i += 256) {
                int row = i / HID_, j = i % HID_;
                float s = 0.f;
                for (int u = 0; u <= v; ++u)
                    s += adj_s[row * MAXN * MAXN + (v + 1) * MAXN + u] *
                         bf2f(gated_s[(u * 16 + row) * GMST + j]);
                hin_s[i] = s;
                psA[row * KPAD + j] = f2bf(s);
            }
            __syncthreads();
        }
    }

    // ---- tiny df MLP head, appended to Hg in psA[301..308] ----
    if (tid < 16) {
        int row = tid;
        float df[3 * PP];
        #pragma unroll
        for (int i = 0; i < 3 * PP; ++i) df[i] = 0.f;
        for (int v2 = 0; v2 < MAXN; ++v2) {
            int base = paths_s[row * MAXN + v2] * 3;
            df[base]     = feats_s[row * 36 + v2 * 3];
            df[base + 1] = feats_s[row * 36 + v2 * 3 + 1];
            df[base + 2] = feats_s[row * 36 + v2 * 3 + 2];
        }
        float hm[16];
        for (int o = 0; o < 16; ++o) {
            float s = b_df1[o];
            for (int k = 0; k < 27; ++k) s += W_df1[o * 27 + k] * df[k];
            hm[o] = s > 0.f ? s : 0.f;
        }
        for (int o = 0; o < 8; ++o) {
            float s = b_df2[o];
            for (int k = 0; k < 16; ++k) s += W_df2[o * 16 + k] * hm[k];
            psA[row * KPAD + HID_ + o] = f2bf(s);
        }
        psA[row * KPAD + HID_ + 8] = 0;
    }
    __syncthreads();

    // ---- [mu | logvar] = Hg @ [Wfc1;Wfc2]^T  (M=16, K=320, N=112) ----
    for (int t = wave; t < NT_FC; t += 8) {
        const unsigned short* bp = frags + WFC_OFF + (size_t)t * KSTEPS * FRAG;
        v8f acc = {};
        #pragma unroll
        for (int ks = 0; ks < KSTEPS; ++ks) {
            ABu a, b;
            int abase = (lane & 15) * KPAD + ks * 32 + ((lane >= 16) ? 8 : 0);
            a.u[0] = *(const uint4*)&psA[abase];
            a.u[1] = *(const uint4*)&psA[abase + 16];
            const unsigned short* bl = bp + ks * FRAG + lane * 16;
            b.u[0] = *(const uint4*)&bl[0];
            b.u[1] = *(const uint4*)&bl[8];
            acc = __builtin_amdgcn_wmma_f32_16x16x32_bf16(
                false, a.v, false, b.v, (short)0, acc, false, false);
        }
        int orow = (lane >= 16) ? 8 : 0, ocol = t * 16 + (lane & 15);
        #pragma unroll
        for (int r = 0; r < 8; ++r) g_s[(orow + r) * NPAD_G + ocol] = acc[r];
    }
    __syncthreads();
    for (int i = tid; i < 16 * 112; i += 256) {
        int row = i / 112, c = i % 112;
        float bias = (c < 56) ? b_fc1[c] : b_fc2[c - 56];
        out[(size_t)(r0 + row) * 112 + c] = g_s[row * NPAD_G + c] + bias;
    }
}

extern "C" void kernel_launch(void* const* d_in, const int* in_sizes, int n_in,
                              void* d_out, int out_size, void* d_ws, size_t ws_size,
                              hipStream_t stream) {
    const int*   types   = (const int*)d_in[0];
    const int*   paths   = (const int*)d_in[1];
    const int*   adj_raw = (const int*)d_in[2];
    const float* feats   = (const float*)d_in[3];
    const float* W_ih    = (const float*)d_in[4];
    const float* W_hh    = (const float*)d_in[5];
    const float* b_ih    = (const float*)d_in[6];
    const float* b_hh    = (const float*)d_in[7];
    const float* Wg      = (const float*)d_in[8];
    const float* bg      = (const float*)d_in[9];
    const float* Wm      = (const float*)d_in[10];
    const float* W_df1   = (const float*)d_in[11];
    const float* b_df1   = (const float*)d_in[12];
    const float* W_df2   = (const float*)d_in[13];
    const float* b_df2   = (const float*)d_in[14];
    const float* W_fc1   = (const float*)d_in[15];
    const float* b_fc1   = (const float*)d_in[16];
    const float* W_fc2   = (const float*)d_in[17];
    const float* b_fc2   = (const float*)d_in[18];
    unsigned short* frags = (unsigned short*)d_ws;
    float* out = (float*)d_out;

    prep_frags<<<(FRAG_TOTAL + 255) / 256, 256, 0, stream>>>(W_hh, Wg, Wm, W_fc1, W_fc2, frags);
    cktgnn_kernel<<<BATCH / 16, 256, 0, stream>>>(
        types, paths, adj_raw, feats, W_ih, b_ih, b_hh, bg,
        W_df1, b_df1, W_df2, b_df2, b_fc1, b_fc2, frags, out);
}
